// WindowAttention_56530359550830
// MI455X (gfx1250) — matrix operand order repaired and checked by reading, not compiled, not run on hardware
//
#include <hip/hip_runtime.h>

typedef __attribute__((ext_vector_type(16))) _Float16 v16h;
typedef __attribute__((ext_vector_type(8)))  float    v8f;

#define WIN_N 64
#define CDIM  256
#define PX    264   // pitch (f16 elems) for [64][256] row-major buffers (pad 8)
#define PV    72    // pitch for Vt [256][64]
#define PP    68    // pitch for P  [64][64]
#define PS    65    // pitch (f32 elems) for logits [64][64]

// ---- WMMA fragment loaders (contiguous k order, same for A and B) ------------------

// A operand: 16x32 f16 tile from row-major [M][K] buffer.
// lane L: M = mBase + L%16 ; VGPR i holds K = kBase + 16*(L/16) + {2i,2i+1}.
__device__ __forceinline__ v16h ldA(const _Float16* buf, int pitch,
                                    int mBase, int kBase, int lane) {
  int m  = mBase + (lane & 15);
  int kb = kBase + ((lane >> 4) << 4);
  const unsigned* p = (const unsigned*)(buf + m * pitch);
  union { v16h v; unsigned u[8]; } r;
#pragma unroll
  for (int i = 0; i < 4; ++i) r.u[i]     = p[(kb >> 1) + i];
#pragma unroll
  for (int i = 0; i < 4; ++i) r.u[4 + i] = p[(kb >> 1) + 4 + i];
  return r.v;
}

// B operand: 32x16 f16 tile (KxN) read from Bt = row-major [N][K] buffer.
// lane L: N = nBase + L%16 ; VGPR i holds K = kBase + 16*(L/16) + {2i,2i+1}.
__device__ __forceinline__ v16h ldB(const _Float16* bt, int pitch,
                                    int nBase, int kBase, int lane) {
  int n  = nBase + (lane & 15);
  int kb = kBase + ((lane >> 4) << 4);
  const unsigned* p = (const unsigned*)(bt + n * pitch);
  union { v16h v; unsigned u[8]; } r;
#pragma unroll
  for (int i = 0; i < 8; ++i) r.u[i] = p[(kb >> 1) + i];
  return r.v;
}

#define WMMA(a, b, c) \
  __builtin_amdgcn_wmma_f32_16x16x32_f16(false, (a), false, (b), (short)0, (c), false, false)

// ---- prep: f32->f16 weight copies + azimuth bias table ---------------------
__global__ void prep_kernel(const float* __restrict__ qkv_w,
                            const float* __restrict__ proj_w,
                            const float* __restrict__ a_p,
                            const float* __restrict__ b_p,
                            const int*   __restrict__ azimuth,
                            _Float16* __restrict__ qkv_w_h,
                            _Float16* __restrict__ proj_w_h,
                            float* __restrict__ bias_phi) {
  int idx = blockIdx.x * blockDim.x + threadIdx.x;
  if (idx < 768 * 256) qkv_w_h[idx]  = (_Float16)qkv_w[idx];
  if (idx < 256 * 256) proj_w_h[idx] = (_Float16)proj_w[idx];
  if (idx < 64 * 64) {
    int az   = azimuth[idx];
    int aidx = az < 0 ? az + 15 : az;
    float azf = (float)az * (6.2831853071795864769f / 64.0f);
    bias_phi[idx] = a_p[aidx] * cosf(azf) + b_p[aidx] * sinf(azf);
  }
}

// ---- fused window attention: one workgroup (8 wave32) per window -----------
__global__ __launch_bounds__(256)
void win_attn_kernel(const float* __restrict__ x,
                     const float* __restrict__ theta_max,
                     const float* __restrict__ qkv_b,
                     const float* __restrict__ proj_b,
                     const float* __restrict__ a_r,
                     const float* __restrict__ b_r,
                     const int*   __restrict__ radius,
                     const _Float16* __restrict__ qkv_w_h,   // [768][256] Bt for QKV
                     const _Float16* __restrict__ proj_w_h,  // [256][256] Bt for proj
                     const float* __restrict__ bias_phi,     // [64][64]
                     float* __restrict__ out) {
  __shared__ _Float16 sX [WIN_N * PX];   // x, later reused for Y = P@V
  __shared__ _Float16 sQ [WIN_N * PX];   // Q * scale
  __shared__ _Float16 sK [WIN_N * PX];   // K (row-major == Bt of K^T)
  __shared__ _Float16 sVt[CDIM  * PV];   // V transposed: [channel][token]
  __shared__ float    sS [WIN_N * PS];   // attention logits (f32)
  __shared__ _Float16 sP [WIN_N * PP];   // softmax probs (f16)

  const int win  = blockIdx.x;
  const int tid  = threadIdx.x;
  const int lane = tid & 31;
  const int wv   = tid >> 5;
  const float tm = theta_max[win >> 6];          // 64 windows per batch sample

  // phase 0: stage x window as f16
  const float* xw = x + (size_t)win * WIN_N * CDIM;
  for (int i = tid; i < WIN_N * CDIM; i += 256) {
    int r = i >> 8, c = i & 255;
    sX[r * PX + c] = (_Float16)xw[i];
  }
  __syncthreads();

  // phase 1: qkv = x @ qkv_w^T + qkv_b  ([64][768]); route to Q/K/Vt in LDS.
  // 4x48 tiles of 16x16; wave w owns 6 N-tiles, all 4 M-tiles (B reuse).
  for (int nt = wv * 6; nt < wv * 6 + 6; ++nt) {
    v8f acc[4];
#pragma unroll
    for (int mt = 0; mt < 4; ++mt) acc[mt] = (v8f)0.0f;
#pragma unroll
    for (int kk = 0; kk < 8; ++kk) {
      v16h b = ldB(qkv_w_h, CDIM, nt * 16, kk * 32, lane);
#pragma unroll
      for (int mt = 0; mt < 4; ++mt) {
        v16h a = ldA(sX, PX, mt * 16, kk * 32, lane);
        acc[mt] = WMMA(a, b, acc[mt]);
      }
    }
    int col = nt * 16 + (lane & 15);
    float bias = qkv_b[col];
#pragma unroll
    for (int mt = 0; mt < 4; ++mt) {
#pragma unroll
      for (int r = 0; r < 8; ++r) {
        int m = mt * 16 + r + ((lane >> 4) << 3);
        float v = acc[mt][r] + bias;
        if (col < 256)      sQ[m * PX + col]           = (_Float16)(v * 0.0625f); // scale=256^-0.5
        else if (col < 512) sK[m * PX + (col - 256)]   = (_Float16)v;
        else                sVt[(col - 512) * PV + m]  = (_Float16)v;             // transposed
      }
    }
  }
  __syncthreads();

  // phase 2: logits = Q@K^T + bias_phi + theta-modulated radius bias
  // 4x4 tiles; wave w owns tiles {w, w+8}
#pragma unroll
  for (int t2 = 0; t2 < 2; ++t2) {
    int t = wv + t2 * 8;
    int mt = t >> 2, nt = t & 3;
    v8f acc = (v8f)0.0f;
#pragma unroll
    for (int kk = 0; kk < 8; ++kk) {
      v16h a = ldA(sQ, PX, mt * 16, kk * 32, lane);
      v16h b = ldB(sK, PX, nt * 16, kk * 32, lane);
      acc = WMMA(a, b, acc);
    }
    int colb = nt * 16 + (lane & 15);
#pragma unroll
    for (int r = 0; r < 8; ++r) {
      int row  = mt * 16 + r + ((lane >> 4) << 3);
      int rad  = radius[row * 64 + colb];
      int ridx = rad < 0 ? rad + 15 : rad;
      float rn = (float)rad * tm * (1.0f / 64.0f);
      float lg = acc[r] + bias_phi[row * 64 + colb]
               + a_r[ridx] * cosf(rn) + b_r[ridx] * sinf(rn);
      sS[row * PS + colb] = lg;
    }
  }
  __syncthreads();

  // phase 3: row softmax (one thread per row), write f16 probs
  if (tid < 64) {
    const float* rowp = &sS[tid * PS];
    float mx = rowp[0];
#pragma unroll 4
    for (int m = 1; m < 64; ++m) mx = fmaxf(mx, rowp[m]);
    float sum = 0.0f;
#pragma unroll 4
    for (int m = 0; m < 64; ++m) sum += __expf(rowp[m] - mx);
    float inv = 1.0f / sum;
#pragma unroll 4
    for (int m = 0; m < 64; ++m)
      sP[tid * PP + m] = (_Float16)(__expf(rowp[m] - mx) * inv);
  }
  __syncthreads();

  // phase 4: Y = P @ V  ([64][256]); 4x16 tiles, wave w owns {w + 8i}. Y -> sX.
#pragma unroll
  for (int i = 0; i < 8; ++i) {
    int t = wv + i * 8;
    int mt = t >> 4, nt = t & 15;
    v8f acc = (v8f)0.0f;
#pragma unroll
    for (int kk = 0; kk < 2; ++kk) {
      v16h a = ldA(sP, PP, mt * 16, kk * 32, lane);
      v16h b = ldB(sVt, PV, nt * 16, kk * 32, lane);
      acc = WMMA(a, b, acc);
    }
    int col = nt * 16 + (lane & 15);
#pragma unroll
    for (int r = 0; r < 8; ++r) {
      int m = mt * 16 + r + ((lane >> 4) << 3);
      sX[m * PX + col] = (_Float16)acc[r];
    }
  }
  __syncthreads();

  // phase 5: out = Y @ proj_w^T + proj_b, stored f32 to HBM
  float* ow = out + (size_t)win * WIN_N * CDIM;
#pragma unroll
  for (int i = 0; i < 8; ++i) {
    int t = wv + i * 8;
    int mt = t >> 4, nt = t & 15;
    v8f acc = (v8f)0.0f;
#pragma unroll
    for (int kk = 0; kk < 8; ++kk) {
      v16h a = ldA(sX, PX, mt * 16, kk * 32, lane);
      v16h b = ldB(proj_w_h, CDIM, nt * 16, kk * 32, lane);
      acc = WMMA(a, b, acc);
    }
    int col = nt * 16 + (lane & 15);
    float bias = proj_b[col];
#pragma unroll
    for (int r = 0; r < 8; ++r) {
      int m = mt * 16 + r + ((lane >> 4) << 3);
      ow[m * CDIM + col] = acc[r] + bias;
    }
  }
}

extern "C" void kernel_launch(void* const* d_in, const int* in_sizes, int n_in,
                              void* d_out, int out_size, void* d_ws, size_t ws_size,
                              hipStream_t stream) {
  (void)in_sizes; (void)n_in; (void)out_size; (void)ws_size;
  const float* x         = (const float*)d_in[0];
  const float* theta_max = (const float*)d_in[1];
  const float* qkv_w     = (const float*)d_in[2];
  const float* qkv_b     = (const float*)d_in[3];
  const float* proj_w    = (const float*)d_in[4];
  const float* proj_b    = (const float*)d_in[5];
  const float* a_p       = (const float*)d_in[6];
  const float* b_p       = (const float*)d_in[7];
  const float* a_r       = (const float*)d_in[8];
  const float* b_r       = (const float*)d_in[9];
  const int*   radius    = (const int*)d_in[10];
  const int*   azimuth   = (const int*)d_in[11];

  char* ws = (char*)d_ws;
  _Float16* qkv_w_h  = (_Float16*)ws;                       // 768*256*2 = 393216 B
  _Float16* proj_w_h = (_Float16*)(ws + 393216);            // 256*256*2 = 131072 B
  float*    bias_phi = (float*)(ws + 393216 + 131072);      // 64*64*4   =  16384 B

  prep_kernel<<<768, 256, 0, stream>>>(qkv_w, proj_w, a_p, b_p, azimuth,
                                       qkv_w_h, proj_w_h, bias_phi);
  win_attn_kernel<<<2048, 256, 0, stream>>>(x, theta_max, qkv_b, proj_b, a_r, b_r,
                                            radius, qkv_w_h, proj_w_h, bias_phi,
                                            (float*)d_out);
}
